// SPAPEAKVAE_49555332661572
// MI455X (gfx1250) — compile-verified, matrix operand order checked by reading, not run on hardware
//
#include <hip/hip_runtime.h>
#include <hip/hip_bf16.h>
#include <math.h>

typedef __attribute__((ext_vector_type(16))) _Float16 v16h;
typedef __attribute__((ext_vector_type(8)))  float    v8f;

// ---------------- workspace layout (float offsets) ----------------
#define OFF_HPRE   0u          // 1024*256 pre-BN encoder+likelihood activations
#define OFF_H      262144u     // 1024*256 post BN+ELU
#define OFF_QMU    524288u     // 1024*10
#define OFF_QVAR   534528u     // 1024*10
#define OFF_LSAMP  544768u     // 1024
#define OFF_KMM    545792u     // 64*64
#define OFF_KMMI   549888u     // 64*64
#define OFF_KNM    553984u     // 1024*64
#define OFF_SIGMA  619520u     // 2 * 64*64
#define OFF_SIGI   627712u     // 2 * 64*64
#define OFF_KMNY   635904u     // 2*64
#define OFF_ALPHA  636032u     // 2*64
#define OFF_GPPM   636160u     // 1024*2
#define OFF_GPPV   638208u     // 1024*2
#define OFF_LAT    640256u     // 1024*10
#define OFF_HDPRE  650496u     // 1024*128
#define OFF_HD     781568u     // 1024*128
#define OFF_ACC    912640u     // 32 scalars (see comments below)
// acc[0]=logdetK  acc[1..2]=logdetSigma_l  acc[3..4]=sum K_tilde
// acc[5..6]=sum trace  acc[7..8]=sum log(noise)  acc[9..10]=sum prec*resid^2
// acc[11]=ce_sum  acc[12]=gauss_KL  acc[13]=recon_loss
// acc[14..15]=L3_l  acc[16..17]=KL_l

#define LOG2PI 1.8378770664093453f

// ---------------- utility ----------------
__global__ void zero_two(float* a, int na, float* b, int nb) {
    int i = blockIdx.x * blockDim.x + threadIdx.x;
    if (i < na) a[i] = 0.f;
    if (i < nb) b[i] = 0.f;
}

// ---------------- fused dual encoder GEMM ----------------
// hpre[1024][256] += y[1024][50000] @ [enc_w1 | l_w1][50000][256]
// grid (16 row-blocks, 25 K-splits), 256 threads (8 waves).
// 50000 = 32*1562 + 16: chunks 0..1561 are full (unguarded b128 staging);
// chunk 1562 is handled once, after the hot loop, by the block that owns it.
#define KCHUNKS 1563
#define KFULL   1562
__global__ void __launch_bounds__(256)
enc_dual_gemm(const float* __restrict__ y,
              const float* __restrict__ w_enc,
              const float* __restrict__ w_l,
              float* __restrict__ hpre) {
    __shared__ __align__(32) _Float16 ldsA[64 * 32];    // 4 KB  (rows x K)
    __shared__ __align__(32) _Float16 ldsB[256 * 32];   // 16 KB (cols x K, transposed)

    const int tid  = threadIdx.x;
    const int wave = tid >> 5, lane = tid & 31;
    const int r0   = blockIdx.x * 64;
    const int per  = (KCHUNKS + gridDim.y - 1) / gridDim.y;
    const int cs   = blockIdx.y * per;
    const int ce   = (cs + per < KCHUNKS) ? (cs + per) : KCHUNKS;
    const int ceF  = (ce < KFULL) ? ce : KFULL;       // full chunks only

    const int rt   = wave & 3;     // row tile within block
    const int half = wave >> 2;    // 0: enc cols, 1: likelihood cols
    const int lrow = lane & 15;
    const int lko  = (lane < 16) ? 0 : 16;

    v8f acc[8];
    for (int t = 0; t < 8; ++t)
        for (int e = 0; e < 8; ++e) acc[t][e] = 0.f;

    // A-fill mapping: thread -> (row, 8-wide K slice)
    const int frow = tid >> 2;
    const int fkk  = (tid & 3) * 8;
    // B-fill mapping: thread t covers 16 contiguous panel floats at t*16
    const int bkk  = (tid * 16) >> 7;   // K row within chunk (0..31)
    const int bcol = (tid * 16) & 127;  // column base (multiple of 16)

    const float* aSrc = y + (size_t)(r0 + frow) * 50000 + fkk;
    _Float16* aDst = &ldsA[frow * 32 + fkk];

    // ---------------- hot loop: full chunks only ----------------
    for (int c = cs; c < ceF; ++c) {
        const int k0 = c * 32;
        {   // stage y tile (f32 -> f16), b128 loads
            const float* src = aSrc + k0;
            __builtin_prefetch(src + 32, 0, 1);
            float4 a0 = ((const float4*)src)[0];
            float4 a1 = ((const float4*)src)[1];
            aDst[0] = (_Float16)a0.x; aDst[1] = (_Float16)a0.y;
            aDst[2] = (_Float16)a0.z; aDst[3] = (_Float16)a0.w;
            aDst[4] = (_Float16)a1.x; aDst[5] = (_Float16)a1.y;
            aDst[6] = (_Float16)a1.z; aDst[7] = (_Float16)a1.w;
        }
        {   // stage weight panels transposed: ldsB[n][k]
            const float* se = w_enc + (size_t)(k0 + bkk) * 128 + bcol;
            const float* sl = w_l   + (size_t)(k0 + bkk) * 128 + bcol;
#pragma unroll
            for (int v = 0; v < 4; ++v) {
                float4 fe = ((const float4*)se)[v];
                float4 fl = ((const float4*)sl)[v];
                int cb = bcol + v * 4;
                ldsB[(cb + 0) * 32 + bkk] = (_Float16)fe.x;
                ldsB[(cb + 1) * 32 + bkk] = (_Float16)fe.y;
                ldsB[(cb + 2) * 32 + bkk] = (_Float16)fe.z;
                ldsB[(cb + 3) * 32 + bkk] = (_Float16)fe.w;
                ldsB[(128 + cb + 0) * 32 + bkk] = (_Float16)fl.x;
                ldsB[(128 + cb + 1) * 32 + bkk] = (_Float16)fl.y;
                ldsB[(128 + cb + 2) * 32 + bkk] = (_Float16)fl.z;
                ldsB[(128 + cb + 3) * 32 + bkk] = (_Float16)fl.w;
            }
        }
        __syncthreads();
        v16h a = *(const v16h*)&ldsA[(rt * 16 + lrow) * 32 + lko];
#pragma unroll
        for (int t = 0; t < 8; ++t) {
            v16h b = *(const v16h*)&ldsB[(half * 128 + t * 16 + lrow) * 32 + lko];
            acc[t] = __builtin_amdgcn_wmma_f32_16x16x32_f16(
                false, a, false, b, (short)0, acc[t], false, false);
        }
        __syncthreads();
    }

    // ---------------- tail chunk (k = 49984..49999 valid) ----------------
    if (ce == KCHUNKS) {
        const int k0 = KFULL * 32;
        {
            const float* src = aSrc + k0;
            for (int q = 0; q < 8; ++q) {
                int k = k0 + fkk + q;
                aDst[q] = (k < 50000) ? (_Float16)src[q] : (_Float16)0.f;
            }
        }
        {
            const bool kin = (k0 + bkk < 50000);   // uniform per thread
            const float* se = w_enc + (size_t)(k0 + bkk) * 128 + bcol;
            const float* sl = w_l   + (size_t)(k0 + bkk) * 128 + bcol;
            for (int e = 0; e < 16; ++e) {
                int cb = bcol + e;
                ldsB[cb * 32 + bkk]         = kin ? (_Float16)se[e] : (_Float16)0.f;
                ldsB[(128 + cb) * 32 + bkk] = kin ? (_Float16)sl[e] : (_Float16)0.f;
            }
        }
        __syncthreads();
        v16h a = *(const v16h*)&ldsA[(rt * 16 + lrow) * 32 + lko];
#pragma unroll
        for (int t = 0; t < 8; ++t) {
            v16h b = *(const v16h*)&ldsB[(half * 128 + t * 16 + lrow) * 32 + lko];
            acc[t] = __builtin_amdgcn_wmma_f32_16x16x32_f16(
                false, a, false, b, (short)0, acc[t], false, false);
        }
        __syncthreads();
    }

    // combine K-split partials
    for (int t = 0; t < 8; ++t) {
        int n = half * 128 + t * 16 + (lane & 15);
#pragma unroll
        for (int i = 0; i < 8; ++i) {
            int m = rt * 16 + i + ((lane < 16) ? 0 : 8);
            atomicAdd(&hpre[(size_t)(r0 + m) * 256 + n], acc[t][i]);
        }
    }
}

// ---------------- batchnorm (population stats over batch) + ELU --------------
__global__ void __launch_bounds__(256)
bn_elu(const float* __restrict__ in, float* __restrict__ out,
       int rows, int cols,
       const float* g_lo, const float* b_lo,
       const float* g_hi, const float* b_hi, int split) {
    __shared__ float red[256];
    const int col = blockIdx.x, tid = threadIdx.x;
    float s = 0.f, s2 = 0.f;
    for (int r = tid; r < rows; r += 256) {
        float v = in[(size_t)r * cols + col];
        s += v; s2 += v * v;
    }
    red[tid] = s; __syncthreads();
    for (int q = 128; q > 0; q >>= 1) { if (tid < q) red[tid] += red[tid + q]; __syncthreads(); }
    float mean = red[0] / rows;
    __syncthreads();
    red[tid] = s2; __syncthreads();
    for (int q = 128; q > 0; q >>= 1) { if (tid < q) red[tid] += red[tid + q]; __syncthreads(); }
    float var = red[0] / rows - mean * mean;
    float g = (col < split) ? g_lo[col] : g_hi[col - split];
    float b = (col < split) ? b_lo[col] : b_hi[col - split];
    float rs = rsqrtf(var + 1e-5f);
    for (int r = tid; r < rows; r += 256) {
        float v = g * (in[(size_t)r * cols + col] - mean) * rs + b;
        out[(size_t)r * cols + col] = (v > 0.f) ? v : expm1f(v);
    }
}

// ---------------- small heads: qnet mu/var, l_samples ----------------
__global__ void __launch_bounds__(256)
heads(const float* __restrict__ h,
      const float* muw, const float* mub,
      const float* vaw, const float* vab,
      const float* lw2, const float* lb2,
      float* qmu, float* qvar, float* lsamp) {
    int r = blockIdx.x * 256 + threadIdx.x;
    if (r >= 1024) return;
    float mu[10], va[10];
    for (int j = 0; j < 10; ++j) { mu[j] = mub[j]; va[j] = vab[j]; }
    const float* hr = h + (size_t)r * 256;
    for (int k = 0; k < 128; ++k) {
        float hv = hr[k];
#pragma unroll
        for (int j = 0; j < 10; ++j) {
            mu[j] += hv * muw[k * 10 + j];
            va[j] += hv * vaw[k * 10 + j];
        }
    }
    for (int j = 0; j < 10; ++j) {
        qmu[r * 10 + j] = mu[j];
        float c = fminf(fmaxf(va[j], -15.f), 15.f);
        qvar[r * 10 + j] = __expf(c);
    }
    float ls = lb2[0];
    for (int k = 0; k < 128; ++k) ls += hr[128 + k] * lw2[k];
    lsamp[r] = ls;
}

// ---------------- Cauchy kernels ----------------
__global__ void build_kmm(const float* ind, float* Kmm) {
    int idx = blockIdx.x * 256 + threadIdx.x;
    if (idx >= 4096) return;
    int i = idx >> 6, j = idx & 63;
    float dx = ind[i * 2] - ind[j * 2], dy = ind[i * 2 + 1] - ind[j * 2 + 1];
    float d2 = dx * dx + dy * dy;
    Kmm[idx] = 1.f / (1.f + d2 / 20.f) + ((i == j) ? 1e-8f : 0.f);
}
__global__ void build_knm(const float* x, const float* ind, float* Knm) {
    int idx = blockIdx.x * 256 + threadIdx.x;
    if (idx >= 65536) return;
    int n = idx >> 6, m = idx & 63;
    float dx = x[n * 2] - ind[m * 2], dy = x[n * 2 + 1] - ind[m * 2 + 1];
    float d2 = dx * dx + dy * dy;
    Knm[idx] = 1.f / (1.f + d2 / 20.f);
}

// ---------------- 64x64 Gauss-Jordan inverse + log|det| ----------------
__global__ void __launch_bounds__(256)
invert64(const float* __restrict__ A, float* __restrict__ Ainv, float* logdet) {
    __shared__ float M[64 * 128];
    __shared__ float fac[64];
    __shared__ float pivv;
    __shared__ float ld;
    const int tid = threadIdx.x;
    for (int idx = tid; idx < 4096; idx += 256) {
        int i = idx >> 6, j = idx & 63;
        M[i * 128 + j] = A[idx];
        M[i * 128 + 64 + j] = (i == j) ? 1.f : 0.f;
    }
    if (tid == 0) ld = 0.f;
    __syncthreads();
    for (int p = 0; p < 64; ++p) {
        if (tid == 0) { pivv = M[p * 128 + p]; ld += __logf(fabsf(pivv)); }
        __syncthreads();
        float inv = 1.f / pivv;
        if (tid < 128) M[p * 128 + tid] *= inv;
        if (tid < 64 && tid != p) fac[tid] = M[tid * 128 + p];
        if (tid == p) fac[tid] = 0.f;
        __syncthreads();
        for (int idx = tid; idx < 64 * 128; idx += 256) {
            int r = idx >> 7, j = idx & 127;
            if (r != p) M[r * 128 + j] -= fac[r] * M[p * 128 + j];
        }
        __syncthreads();
    }
    for (int idx = tid; idx < 4096; idx += 256) {
        int i = idx >> 6, j = idx & 63;
        Ainv[idx] = M[i * 128 + 64 + j];
    }
    if (tid == 0) *logdet = ld;
}

// ---------------- sigma_l = K_mm + ratio * Knm^T diag(prec) Knm --------------
__global__ void __launch_bounds__(256)
build_sigma(const float* Kmm, const float* Knm, const float* qvar, float* sigma) {
    const int l = blockIdx.y;
    const int idx = blockIdx.x * 256 + threadIdx.x;    // 0..4095
    const int i = idx >> 6, j = idx & 63;
    float s = 0.f;
    for (int n = 0; n < 1024; ++n) {
        float prec = 1.f / qvar[n * 10 + l];
        s += Knm[n * 64 + i] * Knm[n * 64 + j] * prec;
    }
    sigma[l * 4096 + idx] = Kmm[idx] + 4.f * s;
}

__global__ void build_kmny(const float* Knm, const float* qmu, const float* qvar, float* kmny) {
    const int l = blockIdx.x, i = threadIdx.x;
    float s = 0.f;
    for (int n = 0; n < 1024; ++n)
        s += Knm[n * 64 + i] * qmu[n * 10 + l] / qvar[n * 10 + l];
    kmny[l * 64 + i] = s;
}

__global__ void build_alpha(const float* siginv, const float* kmny, float* alpha) {
    const int l = blockIdx.x, i = threadIdx.x;
    const float* si = siginv + l * 4096;
    const float* kv = kmny + l * 64;
    float s = 0.f;
    for (int j = 0; j < 64; ++j) s += si[i * 64 + j] * kv[j];
    alpha[l * 64 + i] = s;
}

// ---------------- per-sample GP terms ----------------
__global__ void __launch_bounds__(256)
gp_per_n(const float* __restrict__ Knm, const float* __restrict__ Kmminv,
         const float* __restrict__ siginv, const float* __restrict__ alpha,
         const float* __restrict__ qmu, const float* __restrict__ qvar,
         float* gp_pm, float* gp_pv, float* acc) {
    __shared__ float sKi[4096];
    __shared__ float sSi[4096];
    const int l = blockIdx.y, tid = threadIdx.x;
    for (int idx = tid; idx < 4096; idx += 256) {
        sKi[idx] = Kmminv[idx];
        sSi[idx] = siginv[l * 4096 + idx];
    }
    __syncthreads();
    const int n = blockIdx.x * 256 + tid;
    const float* row = Knm + (size_t)n * 64;
    const float* al  = alpha + l * 64;
    float a1 = 0.f, s1 = 0.f, s2 = 0.f;
    for (int i = 0; i < 64; ++i) {
        float t1 = 0.f, t2 = 0.f;
        for (int j = 0; j < 64; ++j) {
            float rv = row[j];
            t1 += rv * sKi[j * 64 + i];
            t2 += rv * sSi[j * 64 + i];
        }
        float ri = row[i];
        s1 += t1 * ri; s2 += t2 * ri;
        a1 += ri * al[i];
    }
    float meanv = 4.f * a1;               // ratio = N_TRAIN / B = 4
    float var = qvar[n * 10 + l];
    float prec = 1.f / var;
    float pv = 1.f - s1 + s2;
    float mu = qmu[n * 10 + l];
    gp_pm[n * 2 + l] = meanv;
    gp_pv[n * 2 + l] = pv;
    atomicAdd(&acc[3 + l], prec * (1.f - s1));
    atomicAdd(&acc[5 + l], prec * s2);
    atomicAdd(&acc[7 + l], __logf(var));
    atomicAdd(&acc[9 + l], prec * (mu - meanv) * (mu - meanv));
    float ce = -0.5f * (LOG2PI + __logf(var) + (pv + (meanv - mu) * (meanv - mu)) / var);
    atomicAdd(&acc[11], ce);
}

// ---------------- KL & L3 finalization per GP dim ----------------
__global__ void __launch_bounds__(64)
gp_kl(const float* Kmm, const float* siginv, const float* alpha, float* acc) {
    __shared__ float redt[64], redq[64];
    const int l = blockIdx.x, i = threadIdx.x;
    const float* si = siginv + l * 4096;
    const float* al = alpha + l * 64;
    float tr = 0.f, tmp = 0.f;
    for (int j = 0; j < 64; ++j) {
        tr  += si[i * 64 + j] * Kmm[j * 64 + i];    // trace(Sigma^-1 K_mm)
        tmp += Kmm[i * 64 + j] * al[j];
    }
    redt[i] = tr;
    redq[i] = al[i] * tmp;
    __syncthreads();
    for (int s = 32; s > 0; s >>= 1) {
        if (i < s) { redt[i] += redt[i + s]; redq[i] += redq[i + s]; }
        __syncthreads();
    }
    if (i == 0) {
        float trace = redt[0];
        float quad  = 16.f * redq[0];               // ratio^2 * a^T K a
        float ldK = acc[0], ldSig = acc[1 + l];
        // logdet(A_hat) = 2*ldK - ldSig  =>  KL = 0.5*(ldSig - ldK - M + tr + quad)
        acc[16 + l] = 0.5f * (ldSig - ldK - 64.f + trace + quad);
        acc[14 + l] = -0.5f * (acc[3 + l] + acc[5 + l] + acc[7 + l]
                               + 1024.f * LOG2PI + acc[9 + l]);
    }
}

// ---------------- Gaussian latent KL ----------------
__global__ void gauss_kl(const float* qmu, const float* qvar, float* acc12) {
    int idx = blockIdx.x * 256 + threadIdx.x;
    if (idx >= 8192) return;
    int n = idx >> 3, k = 2 + (idx & 7);
    float mu = qmu[n * 10 + k], var = qvar[n * 10 + k];
    atomicAdd(acc12, -0.5f * __logf(var) + 0.5f * (var + mu * mu) - 0.5f);
}

// ---------------- reparameterized latent & decoder hidden ----------------
__global__ void build_latent(const float* gp_pm, const float* gp_pv,
                             const float* qmu, const float* qvar,
                             const float* eps, float* lat) {
    int idx = blockIdx.x * 256 + threadIdx.x;
    if (idx >= 10240) return;
    int n = idx / 10, k = idx % 10;
    float pm, pv;
    if (k < 2) { pm = gp_pm[n * 2 + k]; pv = gp_pv[n * 2 + k]; }
    else       { pm = qmu[idx];         pv = qvar[idx]; }
    lat[idx] = pm + sqrtf(fmaxf(pv, 0.f)) * eps[idx];
}

__global__ void build_hdpre(const float* lat, const float* w, float* hdpre) {
    int idx = blockIdx.x * 256 + threadIdx.x;
    if (idx >= 131072) return;
    int n = idx >> 7, j = idx & 127;
    float s = 0.f;
#pragma unroll
    for (int k = 0; k < 10; ++k) s += lat[n * 10 + k] * w[k * 128 + j];
    hdpre[idx] = s;
}

// ---------------- fused decoder GEMM + sigmoid + Bernoulli recon loss -------
// grid (16 row-blocks x 391 col-blocks), 256 threads (8 waves), K = 128.
__global__ void __launch_bounds__(256)
dec_gemm_recon(const float* __restrict__ hd, const float* __restrict__ w,
               const float* __restrict__ bias, const float* __restrict__ lsamp,
               const float* __restrict__ peak, const float* __restrict__ y,
               float* __restrict__ recon) {
    __shared__ __align__(32) _Float16 ldsA[64 * 128];   // 16 KB
    __shared__ __align__(32) _Float16 ldsB[128 * 128];  // 32 KB
    __shared__ float red[256];
    const int tid = threadIdx.x, wave = tid >> 5, lane = tid & 31;
    const int r0 = blockIdx.x * 64;
    const int c0 = blockIdx.y * 128;
    // stage hd tile (contiguous float4 loads)
    {
        int row = tid >> 2, kk0 = (tid & 3) * 32;
        const float4* src = (const float4*)(hd + (size_t)(r0 + row) * 128 + kk0);
        _Float16* dst = &ldsA[row * 128 + kk0];
#pragma unroll
        for (int v = 0; v < 8; ++v) {
            float4 f = src[v];
            dst[v * 4 + 0] = (_Float16)f.x;
            dst[v * 4 + 1] = (_Float16)f.y;
            dst[v * 4 + 2] = (_Float16)f.z;
            dst[v * 4 + 3] = (_Float16)f.w;
        }
    }
    // stage weight tile transposed: ldsB[nl][k]
    // each thread reads 64 *contiguous* floats along the 50000-wide row
    {
        int k = tid >> 1;
        int cpart = (tid & 1) * 64;
        const float* src = w + (size_t)k * 50000 + c0 + cpart;
        if (c0 + cpart + 64 <= 50000) {
#pragma unroll
            for (int v = 0; v < 16; ++v) {
                float4 f = ((const float4*)src)[v];
                int cb = cpart + v * 4;
                ldsB[(cb + 0) * 128 + k] = (_Float16)f.x;
                ldsB[(cb + 1) * 128 + k] = (_Float16)f.y;
                ldsB[(cb + 2) * 128 + k] = (_Float16)f.z;
                ldsB[(cb + 3) * 128 + k] = (_Float16)f.w;
            }
        } else {
            for (int q = 0; q < 64; ++q) {
                int c = c0 + cpart + q;
                ldsB[(cpart + q) * 128 + k] =
                    (c < 50000) ? (_Float16)src[q] : (_Float16)0.f;
            }
        }
    }
    __syncthreads();
    const int rt = wave & 3, halfc = wave >> 2;
    const int lrow = lane & 15;
    v8f acc[4];
    for (int t = 0; t < 4; ++t)
        for (int e = 0; e < 8; ++e) acc[t][e] = 0.f;
#pragma unroll
    for (int kc = 0; kc < 4; ++kc) {
        int lko = kc * 32 + ((lane < 16) ? 0 : 16);
        v16h a = *(const v16h*)&ldsA[(rt * 16 + lrow) * 128 + lko];
#pragma unroll
        for (int t = 0; t < 4; ++t) {
            v16h b = *(const v16h*)&ldsB[(halfc * 64 + t * 16 + lrow) * 128 + lko];
            acc[t] = __builtin_amdgcn_wmma_f32_16x16x32_f16(
                false, a, false, b, (short)0, acc[t], false, false);
        }
    }
    float loss = 0.f;
    for (int t = 0; t < 4; ++t) {
        int c = c0 + halfc * 64 + t * 16 + lrow;
        if (c < 50000) {
            float pb = 1.f / (1.f + __expf(-peak[c]));
            float bz = bias[c];
#pragma unroll
            for (int i = 0; i < 8; ++i) {
                int m = rt * 16 + i + ((lane < 16) ? 0 : 8);
                int r = r0 + m;
                float pre = acc[t][i] + bz;
                float ms = 1.f / (1.f + __expf(-pre));
                float ls = 1.f / (1.f + __expf(-lsamp[r]));
                float p = ls * ms * pb;
                float yv = y[(size_t)r * 50000 + c];
                float lp = fmaxf(__logf(p), -100.f);
                float l1 = fmaxf(log1pf(-p), -100.f);
                loss -= yv * lp + (1.f - yv) * l1;
            }
        }
    }
    red[tid] = loss;
    __syncthreads();
    for (int s = 128; s > 0; s >>= 1) {
        if (tid < s) red[tid] += red[tid + s];
        __syncthreads();
    }
    if (tid == 0) atomicAdd(recon, red[0]);
}

// ---------------- final ELBO ----------------
__global__ void finalize(const float* acc, float* out) {
    float inside = acc[14] + acc[15] - 0.25f * (acc[16] + acc[17]);
    float gpKL = acc[11] - inside;
    out[0] = acc[13] + 10.f * gpKL + 10.f * acc[12];
}

// ---------------- launcher ----------------
extern "C" void kernel_launch(void* const* d_in, const int* in_sizes, int n_in,
                              void* d_out, int out_size, void* d_ws, size_t ws_size,
                              hipStream_t stream) {
    const float* x          = (const float*)d_in[0];
    const float* y          = (const float*)d_in[1];
    const float* eps        = (const float*)d_in[2];
    const float* inducing   = (const float*)d_in[3];
    const float* enc_w1     = (const float*)d_in[4];
    const float* enc_bn_g   = (const float*)d_in[6];
    const float* enc_bn_b   = (const float*)d_in[7];
    const float* enc_mu_w   = (const float*)d_in[8];
    const float* enc_mu_b   = (const float*)d_in[9];
    const float* enc_var_w  = (const float*)d_in[10];
    const float* enc_var_b  = (const float*)d_in[11];
    const float* dec_w1     = (const float*)d_in[12];
    const float* dec_bn_g   = (const float*)d_in[14];
    const float* dec_bn_b   = (const float*)d_in[15];
    const float* dec_mean_w = (const float*)d_in[16];
    const float* dec_mean_b = (const float*)d_in[17];
    const float* l_w1       = (const float*)d_in[18];
    const float* l_bn_g     = (const float*)d_in[20];
    const float* l_bn_b     = (const float*)d_in[21];
    const float* l_w2       = (const float*)d_in[22];
    const float* l_b2       = (const float*)d_in[23];
    const float* peak_bias  = (const float*)d_in[24];

    float* ws    = (float*)d_ws;
    float* hpre  = ws + OFF_HPRE;
    float* h     = ws + OFF_H;
    float* qmu   = ws + OFF_QMU;
    float* qvar  = ws + OFF_QVAR;
    float* lsamp = ws + OFF_LSAMP;
    float* Kmm   = ws + OFF_KMM;
    float* Kmmi  = ws + OFF_KMMI;
    float* Knm   = ws + OFF_KNM;
    float* sigma = ws + OFF_SIGMA;
    float* sigi  = ws + OFF_SIGI;
    float* kmny  = ws + OFF_KMNY;
    float* alpha = ws + OFF_ALPHA;
    float* gppm  = ws + OFF_GPPM;
    float* gppv  = ws + OFF_GPPV;
    float* lat   = ws + OFF_LAT;
    float* hdpre = ws + OFF_HDPRE;
    float* hd    = ws + OFF_HD;
    float* acc   = ws + OFF_ACC;

    zero_two<<<(262144 + 255) / 256, 256, 0, stream>>>(hpre, 262144, acc, 32);

    enc_dual_gemm<<<dim3(16, 25), 256, 0, stream>>>(y, enc_w1, l_w1, hpre);
    bn_elu<<<256, 256, 0, stream>>>(hpre, h, 1024, 256,
                                    enc_bn_g, enc_bn_b, l_bn_g, l_bn_b, 128);
    heads<<<4, 256, 0, stream>>>(h, enc_mu_w, enc_mu_b, enc_var_w, enc_var_b,
                                 l_w2, l_b2, qmu, qvar, lsamp);

    build_kmm<<<16, 256, 0, stream>>>(inducing, Kmm);
    build_knm<<<256, 256, 0, stream>>>(x, inducing, Knm);
    invert64<<<1, 256, 0, stream>>>(Kmm, Kmmi, acc + 0);
    build_sigma<<<dim3(16, 2), 256, 0, stream>>>(Kmm, Knm, qvar, sigma);
    invert64<<<1, 256, 0, stream>>>(sigma, sigi, acc + 1);
    invert64<<<1, 256, 0, stream>>>(sigma + 4096, sigi + 4096, acc + 2);
    build_kmny<<<2, 64, 0, stream>>>(Knm, qmu, qvar, kmny);
    build_alpha<<<2, 64, 0, stream>>>(sigi, kmny, alpha);
    gp_per_n<<<dim3(4, 2), 256, 0, stream>>>(Knm, Kmmi, sigi, alpha,
                                             qmu, qvar, gppm, gppv, acc);
    gp_kl<<<2, 64, 0, stream>>>(Kmm, sigi, alpha, acc);
    gauss_kl<<<32, 256, 0, stream>>>(qmu, qvar, acc + 12);

    build_latent<<<40, 256, 0, stream>>>(gppm, gppv, qmu, qvar, eps, lat);
    build_hdpre<<<512, 256, 0, stream>>>(lat, dec_w1, hdpre);
    bn_elu<<<128, 256, 0, stream>>>(hdpre, hd, 1024, 128,
                                    dec_bn_g, dec_bn_b, dec_bn_g, dec_bn_b, 128);
    dec_gemm_recon<<<dim3(16, 391), 256, 0, stream>>>(hd, dec_mean_w, dec_mean_b,
                                                      lsamp, peak_bias, y, acc + 13);
    finalize<<<1, 1, 0, stream>>>(acc, (float*)d_out);
}
